// MetaJanusAttn_50440095924318
// MI455X (gfx1250) — matrix operand order repaired and checked by reading, not compiled, not run on hardware
//
#include <hip/hip_runtime.h>
#include <hip/hip_bf16.h>
#include <stdint.h>

// ---------------------------------------------------------------------------
// MetaJanusAttn for MI455X (gfx1250, wave32, WMMA, async LDS copies)
//   echo = x@wjT ; eb = echo@wj ; sc = rowdot(x,eb)/32
//   ja = causal softmax(sc_t*sc_s) ; out = (ja @ (x@wjvT)) @ woT
// GEMM-shaped work runs on v_wmma_f32_16x16x32_bf16 (fp32 accum).
// A-tiles stream via global_load_async_to_lds_b128 (ASYNCcnt path).
// ---------------------------------------------------------------------------

typedef __attribute__((ext_vector_type(16))) __bf16          v16bf;
typedef __attribute__((ext_vector_type(8)))  float           v8f;
typedef __attribute__((ext_vector_type(8)))  unsigned short  us8;

#define LDSTR 40   // padded bf16 row stride in LDS (80B: 16B aligned, bank-spread)

static __device__ __forceinline__ unsigned short f32_to_bf16_rn(float f) {
  union { float f; unsigned u; } v; v.f = f;
  unsigned r = v.u + 0x7FFFu + ((v.u >> 16) & 1u);   // round-to-nearest-even
  return (unsigned short)(r >> 16);
}

union Frag16 { v16bf v; unsigned u[8]; unsigned short h[16]; };

// gfx1250 async copy: global memory -> LDS, 16B per lane, tracked by ASYNCcnt.
static __device__ __forceinline__ void async_copy_b128(uint32_t lds_byte_addr,
                                                       const void* gaddr) {
  asm volatile("global_load_async_to_lds_b128 %0, %1, off"
               :: "v"(lds_byte_addr), "v"((uint64_t)(uintptr_t)gaddr)
               : "memory");
}
static __device__ __forceinline__ void wait_async0() {
#if defined(__has_builtin)
#if __has_builtin(__builtin_amdgcn_s_wait_asynccnt)
  __builtin_amdgcn_s_wait_asynccnt(0);
#else
  asm volatile("s_wait_asynccnt 0x0" ::: "memory");
#endif
#else
  asm volatile("s_wait_asynccnt 0x0" ::: "memory");
#endif
}

// ------------------------- fp32 -> bf16 (flat) ------------------------------
__global__ __launch_bounds__(256)
void f32_to_bf16_kernel(const float* __restrict__ in,
                        unsigned short* __restrict__ out, size_t n) {
  size_t i = ((size_t)blockIdx.x * 256 + threadIdx.x) * 4;
  if (i + 3 < n) {
    float4 v = *(const float4*)(in + i);
    out[i + 0] = f32_to_bf16_rn(v.x);
    out[i + 1] = f32_to_bf16_rn(v.y);
    out[i + 2] = f32_to_bf16_rn(v.z);
    out[i + 3] = f32_to_bf16_rn(v.w);
  }
}

// ------------------- fp32 RxC -> bf16 CxR (tiled transpose) ----------------
__global__ __launch_bounds__(256)
void transpose_to_bf16_kernel(const float* __restrict__ in,
                              unsigned short* __restrict__ out, int R, int C) {
  __shared__ float tile[32][33];
  const int cb = blockIdx.x * 32, rb = blockIdx.y * 32;
  #pragma unroll
  for (int i = 0; i < 32; i += 8)
    tile[threadIdx.y + i][threadIdx.x] =
        in[(size_t)(rb + threadIdx.y + i) * C + cb + threadIdx.x];
  __syncthreads();
  #pragma unroll
  for (int i = 0; i < 32; i += 8)
    out[(size_t)(cb + threadIdx.y + i) * R + rb + threadIdx.x] =
        f32_to_bf16_rn(tile[threadIdx.x][threadIdx.y + i]);
}

// --------------- generic bf16 GEMM:  C(MxN,f32) = A(MxK) @ B(KxN) ----------
// Block: 256 thr (8 waves). Block tile 128x256, K-step 32.
// Wave (wm in 0..1, wn in 0..3) computes 64x64 = 4x4 fragments of 16x16.
// A tile double-buffered in LDS via async copies; B tile reg-staged+transposed.
__global__ __launch_bounds__(256)
void gemm_bf16_wmma_kernel(const unsigned short* __restrict__ A,
                           const unsigned short* __restrict__ B,
                           float* __restrict__ C,
                           int M, int N, int K) {
  __shared__ unsigned short sA [2][128 * LDSTR];  // [buf][row][k]
  __shared__ unsigned short sBt[256 * LDSTR];     // transposed: [n][k]
  (void)M;

  const int tid  = threadIdx.x;
  const int lane = tid & 31;
  const int wave = tid >> 5;
  const int wm   = wave >> 2;          // 0..1  (64-row slab)
  const int wn   = wave & 3;           // 0..3  (64-col slab)
  const int half = lane >> 4;          // 0..1
  const int mq   = lane & 15;          // 0..15

  const int bm0 = blockIdx.y * 128;
  const int bn0 = blockIdx.x * 256;

  const int arow = tid >> 1;           // 0..127
  const int acg  = (tid & 1) * 16;     // 0 / 16
  const int bk   = tid >> 3;           // 0..31
  const int bcg  = (tid & 7) * 32;     // 0..224

  const unsigned short* agp = A + (size_t)(bm0 + arow) * K + acg;
  const uint32_t aLds0 = (uint32_t)(uintptr_t)&sA[0][arow * LDSTR + acg];
  const uint32_t aLds1 = (uint32_t)(uintptr_t)&sA[1][arow * LDSTR + acg];

  // prologue: async-stage A(k=0) into buf0; B(k=0) into registers
  async_copy_b128(aLds0,      agp);
  async_copy_b128(aLds0 + 16, agp + 8);
  us8 rb0, rb1, rb2, rb3;
  {
    const us8* pb = (const us8*)(B + (size_t)bk * N + bn0 + bcg);
    rb0 = pb[0]; rb1 = pb[1]; rb2 = pb[2]; rb3 = pb[3];
  }

  v8f acc[4][4] = {};
  int cur = 0;

  for (int kk = 0; kk < K; kk += 32) {
    // stage B slab (transposed) into LDS
    #pragma unroll
    for (int i = 0; i < 8; ++i) sBt[(bcg + i)      * LDSTR + bk] = rb0[i];
    #pragma unroll
    for (int i = 0; i < 8; ++i) sBt[(bcg + 8 + i)  * LDSTR + bk] = rb1[i];
    #pragma unroll
    for (int i = 0; i < 8; ++i) sBt[(bcg + 16 + i) * LDSTR + bk] = rb2[i];
    #pragma unroll
    for (int i = 0; i < 8; ++i) sBt[(bcg + 24 + i) * LDSTR + bk] = rb3[i];
    wait_async0();           // my async writes into sA[cur] have landed
    __syncthreads();

    if (kk + 32 < K) {       // stream next slab while computing this one
      const unsigned short* ag = agp + (kk + 32);
      const uint32_t al = (cur ? aLds0 : aLds1);
      async_copy_b128(al,      ag);
      async_copy_b128(al + 16, ag + 8);
      const us8* pb = (const us8*)(B + (size_t)(kk + 32 + bk) * N + bn0 + bcg);
      rb0 = pb[0]; rb1 = pb[1]; rb2 = pb[2]; rb3 = pb[3];
      if (kk + 64 < K)
        __builtin_prefetch(B + (size_t)(kk + 64 + bk) * N + bn0 + bcg, 0, 1);
    }

    Frag16 af[4], bf[4];
    #pragma unroll
    for (int mi = 0; mi < 4; ++mi) {
      const unsigned short* base = &sA[cur][(wm * 64 + mi * 16 + mq) * LDSTR];
      #pragma unroll
      for (int v = 0; v < 8; ++v) {
        const int k0 = 2 * v + 16 * half;  // contiguous k order (matches B)
        af[mi].u[v] = *(const unsigned*)&base[k0];
      }
    }
    #pragma unroll
    for (int ni = 0; ni < 4; ++ni) {
      const unsigned short* base = &sBt[(wn * 64 + ni * 16 + mq) * LDSTR + 16 * half];
      #pragma unroll
      for (int v = 0; v < 8; ++v) bf[ni].u[v] = *(const unsigned*)&base[2 * v];
    }
    #pragma unroll
    for (int mi = 0; mi < 4; ++mi)
      #pragma unroll
      for (int ni = 0; ni < 4; ++ni)
        acc[mi][ni] = __builtin_amdgcn_wmma_f32_16x16x32_bf16(
            false, af[mi].v, false, bf[ni].v, (short)0, acc[mi][ni], false, false);

    __syncthreads();
    cur ^= 1;
  }

  #pragma unroll
  for (int mi = 0; mi < 4; ++mi)
    #pragma unroll
    for (int ni = 0; ni < 4; ++ni) {
      const int n = bn0 + wn * 64 + ni * 16 + mq;
      #pragma unroll
      for (int r = 0; r < 8; ++r) {
        const int m = bm0 + wm * 64 + mi * 16 + r + 8 * half;  // ISA C/D layout
        C[(size_t)m * N + n] = acc[mi][ni][r];
      }
    }
}

// -------------------- sc = rowdot(x, eb) * (1/sqrt(E)) ---------------------
__global__ __launch_bounds__(256)
void rowdot_kernel(const float* __restrict__ x, const float* __restrict__ eb,
                   float* __restrict__ sc, int E, float scale) {
  const int row  = blockIdx.x * 8 + (threadIdx.x >> 5);
  const int lane = threadIdx.x & 31;
  const float4* xr = (const float4*)(x  + (size_t)row * E);
  const float4* er = (const float4*)(eb + (size_t)row * E);
  float s = 0.f;
  for (int i = lane; i < E / 4; i += 32) {
    float4 a = xr[i], b = er[i];
    s += a.x * b.x + a.y * b.y + a.z * b.z + a.w * b.w;
  }
  #pragma unroll
  for (int off = 16; off; off >>= 1) s += __shfl_xor(s, off, 32);
  if (lane == 0) sc[row] = s * scale;
}

// --------- per-row softmax stats: m_t = max, zr_t = 1/sum (s <= t) ---------
__global__ __launch_bounds__(256)
void softmax_stats_kernel(const float* __restrict__ sc,
                          float* __restrict__ mrow, float* __restrict__ zrec,
                          int T) {
  const int row  = blockIdx.x * 8 + (threadIdx.x >> 5);
  const int lane = threadIdx.x & 31;
  const int b = row / T, t = row % T;
  const float* scb = sc + (size_t)b * T;
  const float sct = scb[t];
  float mx = -3.4e38f;
  for (int s = lane; s <= t; s += 32) mx = fmaxf(mx, sct * scb[s]);
  #pragma unroll
  for (int off = 16; off; off >>= 1) mx = fmaxf(mx, __shfl_xor(mx, off, 32));
  float z = 0.f;
  for (int s = lane; s <= t; s += 32) z += __expf(sct * scb[s] - mx);
  #pragma unroll
  for (int off = 16; off; off >>= 1) z += __shfl_xor(z, off, 32);
  if (lane == 0) { mrow[row] = mx; zrec[row] = 1.0f / z; }
}

// ------- attention: out_bf16[t,f] = (1/Z_t) * sum_{s<=t} P(t,s) val[s,f] ---
// grid(N/128, T/64, B), block 256. 64 t-rows per block share each val chunk;
// P tiles are built in-register directly in the WMMA A-fragment layout.
__global__ __launch_bounds__(256)
void attn_wmma_kernel(const float* __restrict__ sc,
                      const float* __restrict__ mrow,
                      const float* __restrict__ zrec,
                      const unsigned short* __restrict__ valbf,
                      unsigned short* __restrict__ outbf,
                      int T, int N) {
  __shared__ unsigned short sVt[128 * LDSTR];            // val chunk [f][s]
  __shared__ float s_sc[64], s_m[64], s_zr[64], s_scs[32];

  const int tid = threadIdx.x, lane = tid & 31, wave = tid >> 5;
  const int half = lane >> 4, mq = lane & 15;
  const int b  = blockIdx.z;
  const int t0 = blockIdx.y * 64;
  const int f0 = blockIdx.x * 128;
  const size_t rowbase = (size_t)b * T;

  if (tid < 64) {
    s_sc[tid] = sc  [rowbase + t0 + tid];
    s_m [tid] = mrow[rowbase + t0 + tid];
    s_zr[tid] = zrec[rowbase + t0 + tid];
  }

  const int vk  = tid >> 3;         // 0..31 (s within chunk)
  const int vcg = (tid & 7) * 16;   // f group

  v8f acc[4] = {};
  const int nchunks = (t0 + 63) / 32 + 1;     // causal: only s-chunks <= t0+63
  for (int c = 0; c < nchunks; ++c) {
    const int s0 = c * 32;
    __syncthreads();
    {
      const us8* pv = (const us8*)(valbf + (rowbase + s0 + vk) * N + f0 + vcg);
      us8 v0 = pv[0], v1 = pv[1];
      #pragma unroll
      for (int i = 0; i < 8; ++i) sVt[(vcg + i)     * LDSTR + vk] = v0[i];
      #pragma unroll
      for (int i = 0; i < 8; ++i) sVt[(vcg + 8 + i) * LDSTR + vk] = v1[i];
    }
    if (tid < 32) s_scs[tid] = sc[rowbase + s0 + tid];
    __syncthreads();

    Frag16 vf;                                  // val tile in B-fragment layout
    const unsigned short* vbase = &sVt[(wave * 16 + mq) * LDSTR + 16 * half];
    #pragma unroll
    for (int v = 0; v < 8; ++v) vf.u[v] = *(const unsigned*)&vbase[2 * v];

    #pragma unroll
    for (int mi = 0; mi < 4; ++mi) {
      if (s0 > t0 + mi * 16 + 15) continue;     // wave-uniform: fully masked
      const float sct = s_sc[mi * 16 + mq];
      const float mt  = s_m [mi * 16 + mq];
      const int   trw = t0 + mi * 16 + mq;
      Frag16 pf;                                // P tile in A-fragment layout
      #pragma unroll
      for (int v = 0; v < 8; ++v) {
        const int k0 = 2 * v + 16 * half;
        float p0 = (s0 + k0     <= trw) ? __expf(sct * s_scs[k0]     - mt) : 0.0f;
        float p1 = (s0 + k0 + 1 <= trw) ? __expf(sct * s_scs[k0 + 1] - mt) : 0.0f;
        pf.h[2 * v]     = f32_to_bf16_rn(p0);
        pf.h[2 * v + 1] = f32_to_bf16_rn(p1);
      }
      acc[mi] = __builtin_amdgcn_wmma_f32_16x16x32_bf16(
          false, pf.v, false, vf.v, (short)0, acc[mi], false, false);
    }
  }

  const int n = f0 + wave * 16 + mq;
  #pragma unroll
  for (int mi = 0; mi < 4; ++mi)
    #pragma unroll
    for (int r = 0; r < 8; ++r) {
      const int t = t0 + mi * 16 + r + 8 * half;
      const float o = acc[mi][r] * s_zr[mi * 16 + r + 8 * half];
      outbf[(rowbase + t) * N + n] = f32_to_bf16_rn(o);
    }
}

// ---------------------------------------------------------------------------
extern "C" void kernel_launch(void* const* d_in, const int* in_sizes, int n_in,
                              void* d_out, int out_size, void* d_ws, size_t ws_size,
                              hipStream_t stream) {
  (void)in_sizes; (void)n_in; (void)out_size; (void)ws_size;
  const float* x   = (const float*)d_in[0];   // (B,T,E)
  const float* wj  = (const float*)d_in[1];   // (E,E)
  const float* wjv = (const float*)d_in[2];   // (E,E)
  const float* wo  = (const float*)d_in[3];   // (E,E)

  constexpr int Bb = 4, T = 2048, E = 1024;
  constexpr size_t BT = (size_t)Bb * T;       // 8192 rows

  char* p = (char*)d_ws;
  auto take = [&](size_t bytes) -> char* {
    char* r = p; p += (bytes + 255) & ~(size_t)255; return r;
  };
  unsigned short* xbf  = (unsigned short*)take(BT * E * 2);        // x bf16
  unsigned short* w1   = (unsigned short*)take((size_t)E * E * 2); // wj^T
  unsigned short* w2   = (unsigned short*)take((size_t)E * E * 2); // wj
  unsigned short* w3   = (unsigned short*)take((size_t)E * E * 2); // wjv^T
  unsigned short* w4   = (unsigned short*)take((size_t)E * E * 2); // wo^T
  float*          fbuf = (float*)take(BT * E * 4);                 // echo/eb/val
  unsigned short* bfA  = (unsigned short*)take(BT * E * 2);        // echo/val bf16
  unsigned short* bfB  = (unsigned short*)take(BT * E * 2);        // attn out bf16
  float* scb  = (float*)take(BT * 4);
  float* mrow = (float*)take(BT * 4);
  float* zrec = (float*)take(BT * 4);

  const dim3 blk256(256);
  const dim3 gCvtX((unsigned)((BT * E) / 1024));
  const dim3 gCvtW((unsigned)(((size_t)E * E) / 1024));
  const dim3 gT(E / 32, E / 32), bT(32, 8);
  const dim3 gGemm(E / 256, (unsigned)(BT / 128));
  const dim3 gRows((unsigned)(BT / 8));
  const dim3 gAttn(E / 128, T / 64, Bb);

  // operand precision conversion
  f32_to_bf16_kernel<<<gCvtX, blk256, 0, stream>>>(x, xbf, BT * E);
  transpose_to_bf16_kernel<<<gT, bT, 0, stream>>>(wj,  w1, E, E);
  f32_to_bf16_kernel<<<gCvtW, blk256, 0, stream>>>(wj, w2, (size_t)E * E);
  transpose_to_bf16_kernel<<<gT, bT, 0, stream>>>(wjv, w3, E, E);
  transpose_to_bf16_kernel<<<gT, bT, 0, stream>>>(wo,  w4, E, E);

  // echo = x @ wj^T
  gemm_bf16_wmma_kernel<<<gGemm, blk256, 0, stream>>>(xbf, w1, fbuf, (int)BT, E, E);
  f32_to_bf16_kernel<<<gCvtX, blk256, 0, stream>>>(fbuf, bfA, BT * E);
  // eb = echo @ wj
  gemm_bf16_wmma_kernel<<<gGemm, blk256, 0, stream>>>(bfA, w2, fbuf, (int)BT, E, E);
  // sc = rowdot(x, eb) / sqrt(E)
  rowdot_kernel<<<gRows, blk256, 0, stream>>>(x, fbuf, scb, E, 1.0f / 32.0f);
  // val = x @ wjv^T
  gemm_bf16_wmma_kernel<<<gGemm, blk256, 0, stream>>>(xbf, w3, fbuf, (int)BT, E, E);
  f32_to_bf16_kernel<<<gCvtX, blk256, 0, stream>>>(fbuf, bfA, BT * E);
  // softmax stats, then attention contraction (WMMA)
  softmax_stats_kernel<<<gRows, blk256, 0, stream>>>(scb, mrow, zrec, T);
  attn_wmma_kernel<<<gAttn, blk256, 0, stream>>>(scb, mrow, zrec, bfA, bfB, T, E);
  // out = attn @ wo^T
  gemm_bf16_wmma_kernel<<<gGemm, blk256, 0, stream>>>(bfB, w4, (float*)d_out,
                                                      (int)BT, E, E);
}
